// Block_81054622810301
// MI455X (gfx1250) — compile-verified, matrix operand order checked by reading, not run on hardware
//
#include <hip/hip_runtime.h>

// ---------------------------------------------------------------------------
// Transformer encoder block forward for gfx1250 (MI455X), wave32 + WMMA.
// B=8, S=1024, D=768, H=12, DH=64, FF=3072.
// Outputs: x_out [B,S,D] f32 followed by attention probs [B,H,S,S] f32.
//
// All GEMM operands are staged as f16 (WMMA truncates to f16 anyway):
//   - weights are transpose-converted once to f16 [N][K] (k-contiguous rows)
//   - activations are written as f16 by LN / GEMM epilogues
//   - V is written transposed [B,D,S] so P@V's B operand is n-major too
// Tile staging is then a pure 16B-chunk copy -> GLOBAL_LOAD_ASYNC_TO_LDS_B128
// (ASYNCcnt) when the toolchain exposes it; sync b128+ds_store_b128 fallback.
// ---------------------------------------------------------------------------

typedef __attribute__((ext_vector_type(16))) _Float16 v16h;
typedef __attribute__((ext_vector_type(8)))  float    v8f;
typedef __attribute__((ext_vector_type(4)))  int      v4i;

#define TB_B 8
#define TB_S 1024
#define TB_D 768
#define TB_H 12
#define TB_DH 64
#define TB_FF 3072

#define BM  128
#define KT  32
#define KTP 40   // 80B rows: 16B aligned, 20-dword stride -> conflict-free

#define FLAG_BIAS  1
#define FLAG_GELU  2
#define FLAG_RESID 4

union Frag16 { unsigned u[8]; v16h h; };
union H2Pack { _Float16 h[2]; unsigned u; };

#if defined(__has_builtin)
#if __has_builtin(__builtin_amdgcn_global_load_async_to_lds_b128) && \
    __has_builtin(__builtin_amdgcn_s_wait_asynccnt)
#define USE_ASYNC_LDS 1
#endif
#endif
#ifndef USE_ASYNC_LDS
#define USE_ASYNC_LDS 0
#endif

#if USE_ASYNC_LDS
typedef v4i __attribute__((address_space(1)))* g128_t;   // global int4*
typedef v4i __attribute__((address_space(3)))* l128_t;   // LDS int4*
#endif

// ---------------------------------------------------------------------------
// Generic batched GEMM:  C = epilogue( scale * A @ B ) via v_wmma_f32_16x16x32_f16
//   A[m,k] at A + m*lda + k            (k-contiguous; f16 if A16 else f32)
//   B[n,k] at B + n*ldbRow + k         (f16, n-major, k-contiguous)
//   C[m,n] at C + m*ldcM + n*ldcN      (f16 if CHALF else f32; strided store
//                                       mode gives transposed outputs)
//   batch z = zo*innerCount + zi with element offsets per operand.
// Block: 256 threads = 8 waves (4x2). Tile BM x (32*NT); wave: 2 x NT WMMAs.
// LDS ping-pong double buffering, copies for tile i+1 in flight during WMMAs.
// ---------------------------------------------------------------------------
template<int NT, bool A16, bool CHALF>
__global__ __launch_bounds__(256) void gemm_wmma(
    const void* __restrict__ Avp, const _Float16* __restrict__ Bh,
    const float* __restrict__ bias, const float* __restrict__ R,
    void* __restrict__ Cvp,
    int M, int N, int K,
    int lda, int ldbRow, long long ldcM, long long ldcN,
    int innerCount,
    long long aOuter, long long aInner,
    long long bOuter, long long bInner,
    long long cOuter, long long cInner,
    float scale, int flags)
{
    constexpr int BN  = 32 * NT;      // 128 (NT=4) or 64 (NT=2)
    constexpr int ACH = BM / 64;      // 16B chunks per thread for A (f16) = 2
    constexpr int BCH = BN / 64;      // 16B chunks per thread for B = 2 or 1

    __shared__ __align__(16) _Float16 As[2][BM][KTP];
    __shared__ __align__(16) _Float16 Bs[2][BN][KTP];

    const int z  = blockIdx.z;
    const int zo = z / innerCount;
    const int zi = z - zo * innerCount;
    const _Float16* Ah = A16 ? (const _Float16*)Avp + zo * aOuter + zi * aInner : nullptr;
    const float*    Af = A16 ? nullptr : (const float*)Avp + zo * aOuter + zi * aInner;
    Bh += zo * bOuter + zi * bInner;
    const long long coff = zo * cOuter + zi * cInner;

    const int m0   = blockIdx.x * BM;
    const int n0   = blockIdx.y * BN;
    const int tid  = threadIdx.x;
    const int wid  = tid >> 5;
    const int lane = tid & 31;
    const int wm   = wid >> 1;    // 0..3 : 32-row strip
    const int wn   = wid & 1;     // 0..1 : (16*NT)-col strip
    const int lrow = lane & 15;
    const int lhi  = lane >> 4;   // 0/1

#if !USE_ASYNC_LDS
    uint4  aS[ACH > 0 ? ACH : 1];
    float4 aF[4];
    uint4  bS[BCH > 0 ? BCH : 1];
#else
    float4 aF[4];                 // only used when !A16
#endif

    // ---- stage issue: start global->LDS traffic for buffer `buf`, k=kbase
    auto stage_issue = [&](int buf, int kbase) {
        if (A16) {
#if USE_ASYNC_LDS
            #pragma unroll
            for (int j = 0; j < ACH; ++j) {
                const int c = tid + j * 256, row = c >> 2, cc = c & 3;
                const _Float16* g = Ah + (long long)(m0 + row) * lda + kbase + cc * 8;
                __builtin_amdgcn_global_load_async_to_lds_b128(
                    (g128_t)g, (l128_t)&As[buf][row][cc * 8], 0, 0);
            }
#else
            #pragma unroll
            for (int j = 0; j < ACH; ++j) {
                const int c = tid + j * 256, row = c >> 2, cc = c & 3;
                aS[j] = *(const uint4*)(Ah + (long long)(m0 + row) * lda + kbase + cc * 8);
            }
#endif
        } else {
            // f32 A: thread -> (row = tid/2, half = tid&1), 16 contiguous floats
            const int row = tid >> 1, half = tid & 1;
            #pragma unroll
            for (int j = 0; j < 4; ++j)
                aF[j] = *(const float4*)(Af + (long long)(m0 + row) * lda +
                                         kbase + half * 16 + j * 4);
        }
#if USE_ASYNC_LDS
        #pragma unroll
        for (int j = 0; j < BCH; ++j) {
            const int c = tid + j * 256, row = c >> 2, cc = c & 3;
            const _Float16* g = Bh + (long long)(n0 + row) * ldbRow + kbase + cc * 8;
            __builtin_amdgcn_global_load_async_to_lds_b128(
                (g128_t)g, (l128_t)&Bs[buf][row][cc * 8], 0, 0);
        }
#else
        #pragma unroll
        for (int j = 0; j < BCH; ++j) {
            const int c = tid + j * 256, row = c >> 2, cc = c & 3;
            bS[j] = *(const uint4*)(Bh + (long long)(n0 + row) * ldbRow + kbase + cc * 8);
        }
#endif
    };

    // ---- stage commit: finish writes into LDS buffer `buf`
    auto stage_commit = [&](int buf) {
        if (!A16) {
            const int row = tid >> 1, half = tid & 1;
            unsigned u[8];
            #pragma unroll
            for (int j = 0; j < 4; ++j) {
                H2Pack p0, p1;
                p0.h[0] = (_Float16)aF[j].x; p0.h[1] = (_Float16)aF[j].y;
                p1.h[0] = (_Float16)aF[j].z; p1.h[1] = (_Float16)aF[j].w;
                u[2 * j] = p0.u; u[2 * j + 1] = p1.u;
            }
            uint4* d = (uint4*)&As[buf][row][half * 16];
            d[0] = make_uint4(u[0], u[1], u[2], u[3]);
            d[1] = make_uint4(u[4], u[5], u[6], u[7]);
        }
#if !USE_ASYNC_LDS
        if (A16) {
            #pragma unroll
            for (int j = 0; j < ACH; ++j) {
                const int c = tid + j * 256, row = c >> 2, cc = c & 3;
                *(uint4*)&As[buf][row][cc * 8] = aS[j];
            }
        }
        #pragma unroll
        for (int j = 0; j < BCH; ++j) {
            const int c = tid + j * 256, row = c >> 2, cc = c & 3;
            *(uint4*)&Bs[buf][row][cc * 8] = bS[j];
        }
#else
        __builtin_amdgcn_s_wait_asynccnt(0);
#endif
    };

    v8f acc[2][NT] = {};

    auto compute = [&](int buf) {
        Frag16 fa[2], fb[NT];
        #pragma unroll
        for (int t = 0; t < 2; ++t) {
            const uint4* pa = (const uint4*)&As[buf][wm * 32 + t * 16 + lrow][0];
            const uint4 lo = pa[lhi], hi = pa[2 + lhi];
            fa[t].u[0] = lo.x; fa[t].u[1] = lo.y; fa[t].u[2] = lo.z; fa[t].u[3] = lo.w;
            fa[t].u[4] = hi.x; fa[t].u[5] = hi.y; fa[t].u[6] = hi.z; fa[t].u[7] = hi.w;
        }
        #pragma unroll
        for (int t = 0; t < NT; ++t) {
            const uint4* pb = (const uint4*)&Bs[buf][wn * (16 * NT) + t * 16 + lrow][0];
            const uint4 lo = pb[lhi], hi = pb[2 + lhi];
            fb[t].u[0] = lo.x; fb[t].u[1] = lo.y; fb[t].u[2] = lo.z; fb[t].u[3] = lo.w;
            fb[t].u[4] = hi.x; fb[t].u[5] = hi.y; fb[t].u[6] = hi.z; fb[t].u[7] = hi.w;
        }
        #pragma unroll
        for (int tm = 0; tm < 2; ++tm)
            #pragma unroll
            for (int tn = 0; tn < NT; ++tn)
                acc[tm][tn] = __builtin_amdgcn_wmma_f32_16x16x32_f16(
                    false, fa[tm].h, false, fb[tn].h,
                    (short)0, acc[tm][tn], false, false);
    };

    // ---- pipelined main loop (LDS ping-pong) ----
    stage_issue(0, 0);
    stage_commit(0);
    __syncthreads();

    int buf = 0;
    for (int k0 = 0; k0 < K; k0 += KT) {
        const bool more = (k0 + KT) < K;
        if (more) stage_issue(buf ^ 1, k0 + KT);   // copies in flight
        compute(buf);                              // WMMAs from LDS[buf]
        if (more) stage_commit(buf ^ 1);
        __syncthreads();
        buf ^= 1;
    }

    // ---- epilogue: C/D layout VGPR r -> M = r + 8*lhi, N = lrow ----
    #pragma unroll
    for (int tm = 0; tm < 2; ++tm) {
        #pragma unroll
        for (int tn = 0; tn < NT; ++tn) {
            const int n = n0 + wn * (16 * NT) + tn * 16 + lrow;
            #pragma unroll
            for (int r = 0; r < 8; ++r) {
                const int m = m0 + wm * 32 + tm * 16 + r + (lhi << 3);
                float val = acc[tm][tn][r] * scale;
                if (flags & FLAG_BIAS)  val += bias[n];
                if (flags & FLAG_GELU)  val = 0.5f * val * (1.0f + erff(val * 0.70710678118f));
                const long long idx = coff + (long long)m * ldcM + (long long)n * ldcN;
                if (flags & FLAG_RESID) val += R[idx];
                if (CHALF) ((_Float16*)Cvp)[idx] = (_Float16)val;
                else       ((float*)Cvp)[idx]    = val;
            }
        }
    }
}

// ---------------------------------------------------------------------------
// Transpose + f32->f16 convert:  wt[n*K + k] = (f16) w[k*N + n]
// ---------------------------------------------------------------------------
__global__ __launch_bounds__(256) void transpose_cvt_k(
    const float* __restrict__ w, _Float16* __restrict__ wt, int K, int N)
{
    __shared__ float t[32][33];
    const int n0 = blockIdx.x * 32, k0 = blockIdx.y * 32;
    const int tx = threadIdx.x & 31, ty = threadIdx.x >> 5;  // 32 x 8
    #pragma unroll
    for (int j = 0; j < 32; j += 8)
        t[ty + j][tx] = w[(long long)(k0 + ty + j) * N + (n0 + tx)];
    __syncthreads();
    #pragma unroll
    for (int j = 0; j < 32; j += 8)
        wt[(long long)(n0 + ty + j) * K + (k0 + tx)] = (_Float16)t[tx][ty + j];
}

// ---------------------------------------------------------------------------
// LayerNorm over last dim (Dn=768), one block per row, f16 output.
// ---------------------------------------------------------------------------
__global__ __launch_bounds__(256) void layernorm_k(
    const float* __restrict__ x, const float* __restrict__ g,
    const float* __restrict__ b, _Float16* __restrict__ y, int Dn)
{
    __shared__ float sm[256];
    const long long row = blockIdx.x;
    const float* xr = x + row * Dn;
    const int tid = threadIdx.x;

    float s = 0.f;
    for (int i = tid; i < Dn; i += 256) s += xr[i];
    sm[tid] = s; __syncthreads();
    for (int off = 128; off > 0; off >>= 1) {
        if (tid < off) sm[tid] += sm[tid + off];
        __syncthreads();
    }
    const float mu = sm[0] / (float)Dn;
    __syncthreads();

    float vs = 0.f;
    for (int i = tid; i < Dn; i += 256) { float d = xr[i] - mu; vs += d * d; }
    sm[tid] = vs; __syncthreads();
    for (int off = 128; off > 0; off >>= 1) {
        if (tid < off) sm[tid] += sm[tid + off];
        __syncthreads();
    }
    const float rstd = rsqrtf(sm[0] / (float)Dn + 1e-6f);

    for (int i = tid; i < Dn; i += 256)
        y[row * Dn + i] = (_Float16)((xr[i] - mu) * rstd * g[i] + b[i]);
}

// ---------------------------------------------------------------------------
// In-place softmax over rows of length n (=1024), one block per row (f32).
// ---------------------------------------------------------------------------
__global__ __launch_bounds__(256) void softmax_k(float* __restrict__ p, int n)
{
    __shared__ float sm[256];
    const long long row = blockIdx.x;
    float* pr = p + row * (long long)n;
    const int tid = threadIdx.x;

    float mx = -3.4e38f;
    for (int i = tid; i < n; i += 256) mx = fmaxf(mx, pr[i]);
    sm[tid] = mx; __syncthreads();
    for (int off = 128; off > 0; off >>= 1) {
        if (tid < off) sm[tid] = fmaxf(sm[tid], sm[tid + off]);
        __syncthreads();
    }
    mx = sm[0]; __syncthreads();

    float s = 0.f;
    for (int i = tid; i < n; i += 256) {
        const float e = expf(pr[i] - mx);
        pr[i] = e;
        s += e;
    }
    sm[tid] = s; __syncthreads();
    for (int off = 128; off > 0; off >>= 1) {
        if (tid < off) sm[tid] += sm[tid + off];
        __syncthreads();
    }
    const float inv = 1.0f / sm[0];
    for (int i = tid; i < n; i += 256) pr[i] *= inv;
}

// ---------------------------------------------------------------------------
extern "C" void kernel_launch(void* const* d_in, const int* in_sizes, int n_in,
                              void* d_out, int out_size, void* d_ws, size_t ws_size,
                              hipStream_t stream)
{
    (void)in_sizes; (void)n_in; (void)out_size; (void)ws_size;

    const float* x    = (const float*)d_in[0];
    const float* wq   = (const float*)d_in[1];
    const float* bq   = (const float*)d_in[2];
    const float* wk   = (const float*)d_in[3];
    const float* bk   = (const float*)d_in[4];
    const float* wv   = (const float*)d_in[5];
    const float* bv   = (const float*)d_in[6];
    const float* wo   = (const float*)d_in[7];
    const float* bo   = (const float*)d_in[8];
    const float* w1   = (const float*)d_in[9];
    const float* b1   = (const float*)d_in[10];
    const float* w2   = (const float*)d_in[11];
    const float* b2   = (const float*)d_in[12];
    const float* ln1g = (const float*)d_in[13];
    const float* ln1b = (const float*)d_in[14];
    const float* ln2g = (const float*)d_in[15];
    const float* ln2b = (const float*)d_in[16];

    const long long BSD  = (long long)TB_B * TB_S * TB_D;   // 6,291,456
    const long long SS   = (long long)TB_S * TB_S;          // 1,048,576
    const long long SD   = (long long)TB_S * TB_D;
    const long long DD   = (long long)TB_D * TB_D;          // 589,824
    const long long DFF  = (long long)TB_D * TB_FF;         // 2,359,296
    const long long BSFF = (long long)TB_B * TB_S * TB_FF;  // 25,165,824

    // ---- workspace carve: f32 first (alignment), then f16 ----
    float* x1 = (float*)d_ws;                 // [B,S,D] residual stream (f32)
    _Float16* hb = (_Float16*)(x1 + BSD);
    _Float16* xn16  = hb;  hb += BSD;         // LN1 out (reused as LN2 out)
    _Float16* q16   = hb;  hb += BSD;
    _Float16* k16   = hb;  hb += BSD;
    _Float16* v16t  = hb;  hb += BSD;         // V transposed [B, D, S]
    _Float16* ctx16 = hb;  hb += BSD;
    _Float16* ffh16 = hb;  hb += BSFF;
    _Float16* wqT   = hb;  hb += DD;          // [N=768][K=768]
    _Float16* wkT   = hb;  hb += DD;
    _Float16* wvT   = hb;  hb += DD;
    _Float16* woT   = hb;  hb += DD;
    _Float16* w1T   = hb;  hb += DFF;         // [N=3072][K=768]
    _Float16* w2T   = hb;  hb += DFF;         // [N=768][K=3072]

    float* xout  = (float*)d_out;             // [B,S,D]
    float* probs = xout + BSD;                // [B,H,S,S]

    const dim3 blk(256);
    const int Mrows = TB_B * TB_S;            // 8192

    // 0) weight transpose+convert (once per call)
    transpose_cvt_k<<<dim3(TB_D / 32, TB_D / 32), blk, 0, stream>>>(wq, wqT, TB_D, TB_D);
    transpose_cvt_k<<<dim3(TB_D / 32, TB_D / 32), blk, 0, stream>>>(wk, wkT, TB_D, TB_D);
    transpose_cvt_k<<<dim3(TB_D / 32, TB_D / 32), blk, 0, stream>>>(wv, wvT, TB_D, TB_D);
    transpose_cvt_k<<<dim3(TB_D / 32, TB_D / 32), blk, 0, stream>>>(wo, woT, TB_D, TB_D);
    transpose_cvt_k<<<dim3(TB_FF / 32, TB_D / 32), blk, 0, stream>>>(w1, w1T, TB_D, TB_FF);
    transpose_cvt_k<<<dim3(TB_D / 32, TB_FF / 32), blk, 0, stream>>>(w2, w2T, TB_FF, TB_D);

    // 1) LN1 -> xn16
    layernorm_k<<<Mrows, blk, 0, stream>>>(x, ln1g, ln1b, xn16, TB_D);

    // 2) Q, K projections (f16 out), V projection with transposed store
    const dim3 gProj(Mrows / BM, TB_D / 128, 1);
    gemm_wmma<4, true, true><<<gProj, blk, 0, stream>>>(xn16, wqT, bq, nullptr, q16,
        Mrows, TB_D, TB_D, TB_D, TB_D, TB_D, 1,
        1, 0, 0, 0, 0, 0, 0, 1.0f, FLAG_BIAS);
    gemm_wmma<4, true, true><<<gProj, blk, 0, stream>>>(xn16, wkT, bk, nullptr, k16,
        Mrows, TB_D, TB_D, TB_D, TB_D, TB_D, 1,
        1, 0, 0, 0, 0, 0, 0, 1.0f, FLAG_BIAS);
    const dim3 gV(TB_S / BM, TB_D / 128, TB_B);       // per-batch: v16t[b][n][s]
    gemm_wmma<4, true, true><<<gV, blk, 0, stream>>>(xn16, wvT, bv, nullptr, v16t,
        TB_S, TB_D, TB_D, TB_D, TB_D, /*ldcM=*/1, /*ldcN=*/TB_S,
        1, SD, 0, 0, 0, (long long)TB_D * TB_S, 0, 1.0f, FLAG_BIAS);

    // 3) scores[b,h] = (q_h @ k_h^T)/8 -> probs (f32, written once)
    const dim3 gScore(TB_S / BM, TB_S / 128, TB_B * TB_H);
    gemm_wmma<4, true, false><<<gScore, blk, 0, stream>>>(q16, k16, nullptr, nullptr, probs,
        TB_S, TB_S, TB_DH, TB_D, TB_D, TB_S, 1,
        TB_H, SD, TB_DH, SD, TB_DH,
        (long long)TB_H * SS, SS, 0.125f, 0);

    // 4) softmax in place on probs
    softmax_k<<<TB_B * TB_H * TB_S, blk, 0, stream>>>(probs, TB_S);

    // 5) ctx[b,h] = probs @ v_h  (A = f32 probs, B = v16t) -> ctx16[b,s,h*64+d]
    const dim3 gCtx(TB_S / BM, TB_DH / 64, TB_B * TB_H);
    gemm_wmma<2, false, true><<<gCtx, blk, 0, stream>>>(probs, v16t, nullptr, nullptr, ctx16,
        TB_S, TB_DH, TB_S, TB_S, /*ldbRow=*/TB_S, TB_D, 1,
        TB_H, (long long)TB_H * SS, SS,
        (long long)TB_D * TB_S, (long long)TB_DH * TB_S,
        SD, TB_DH, 1.0f, 0);

    // 6) attn out + residual: x1 = ctx @ wo + bo + x   (f32 out)
    gemm_wmma<4, true, false><<<gProj, blk, 0, stream>>>(ctx16, woT, bo, x, x1,
        Mrows, TB_D, TB_D, TB_D, TB_D, TB_D, 1,
        1, 0, 0, 0, 0, 0, 0, 1.0f, FLAG_BIAS | FLAG_RESID);

    // 7) LN2 -> xn16 (buffer reuse)
    layernorm_k<<<Mrows, blk, 0, stream>>>(x1, ln2g, ln2b, xn16, TB_D);

    // 8) FF1 + exact GELU: ffh16 = gelu(xn @ w1 + b1)
    const dim3 gFF1(Mrows / BM, TB_FF / 128, 1);
    gemm_wmma<4, true, true><<<gFF1, blk, 0, stream>>>(xn16, w1T, b1, nullptr, ffh16,
        Mrows, TB_FF, TB_D, TB_D, TB_D, TB_FF, 1,
        1, 0, 0, 0, 0, 0, 0, 1.0f, FLAG_BIAS | FLAG_GELU);

    // 9) FF2 + residual: xout = ffh @ w2 + b2 + x1   (f32 out)
    gemm_wmma<4, true, false><<<gProj, blk, 0, stream>>>(ffh16, w2T, b2, x1, xout,
        Mrows, TB_D, TB_FF, TB_FF, TB_FF, TB_D, 1,
        1, 0, 0, 0, 0, 0, 0, 1.0f, FLAG_BIAS | FLAG_RESID);
}
